// QKVBlock_74569222193163
// MI455X (gfx1250) — matrix operand order repaired and checked by reading, not compiled, not run on hardware
//
#include <hip/hip_runtime.h>
#include <hip/hip_bf16.h>
#include <cstdint>

// Problem constants (fixed by the reference)
#define B_  2
#define S_  2048
#define E_  2048
#define H_  16
#define D_  128
#define M_  (B_ * S_)          // 4096 rows for projections
#define SCALE_ 0.08838834764831845f   // 1/sqrt(128)

typedef __attribute__((ext_vector_type(16))) __bf16 v16bf;
typedef __attribute__((ext_vector_type(8)))  __bf16 v8bf;
typedef __attribute__((ext_vector_type(8)))  float  v8f;
typedef __attribute__((ext_vector_type(4)))  unsigned int u32x4;
typedef __attribute__((ext_vector_type(8)))  unsigned int u32x8;

static __device__ inline v16bf cat8(v8bf a, v8bf b) {
  return __builtin_shufflevector(a, b, 0,1,2,3,4,5,6,7,8,9,10,11,12,13,14,15);
}

static __device__ inline v8f wmma_bf16(v16bf a, v16bf b, v8f c) {
  // D = A(16x32) * B(32x16) + C, f32 accumulate
  return __builtin_amdgcn_wmma_f32_16x16x32_bf16(
      /*neg_a=*/false, a, /*neg_b=*/false, b,
      /*c_mod=*/(short)0, c, /*reuse_a=*/false, /*reuse_b=*/false);
}

// Async global->LDS copy (CDNA5): data bypasses VGPRs, tracked by ASYNCcnt.
static __device__ inline void async_ld_b128(uint32_t lds_addr, uint64_t gaddr) {
  asm volatile("global_load_async_to_lds_b128 %0, %1, off"
               :: "v"(lds_addr), "v"(gaddr) : "memory");
}
static __device__ inline void wait_async0() {
  asm volatile("s_wait_asynccnt 0x0" ::: "memory");
}

// ---------------------------------------------------------------- convert
// 4 f32 -> 4 bf16 per thread
__global__ void cvt_f32_bf16x4(const float4* __restrict__ in,
                               uint2* __restrict__ out, int n4) {
  int i = blockIdx.x * blockDim.x + threadIdx.x;
  if (i >= n4) return;
  float4 f = in[i];
  union { __bf16 h[4]; uint2 u; } c;
  c.h[0] = (__bf16)f.x; c.h[1] = (__bf16)f.y;
  c.h[2] = (__bf16)f.z; c.h[3] = (__bf16)f.w;
  out[i] = c.u;
}

// ---------------------------------------------------------------- GEMM
// C[M,N] = A[M,K] @ W[N,K]^T + bias[N]
// K-step 64, double-buffered LDS, async global->LDS staging.
// mode 0: scatter bf16 to [B,H,S,D]   mode 1: f32 to out[row*N+col]
__global__ __launch_bounds__(256) void gemm_bf16(
    const __bf16* __restrict__ A, const __bf16* __restrict__ W,
    const float* __restrict__ bias,
    __bf16* __restrict__ outb, float* __restrict__ outf, int mode)
{
  const int Kdim = E_, Ndim = E_;
  __shared__ __align__(128) __bf16 At[2][128][64];   // 2 x 16 KB
  __shared__ __align__(128) __bf16 Wt[2][128][64];   // 2 x 16 KB

  int t    = threadIdx.x;
  int lane = t & 31, wave = t >> 5;
  int l16  = lane & 15, hs = lane >> 4;      // half-wave select
  int M0 = blockIdx.y * 128, N0 = blockIdx.x * 128;

  // staging geometry: 2 threads per row, 32 elements (64 B) each
  int sr = t >> 1;
  int sh = (t & 1) * 32;
  const __bf16* Arow = A + (size_t)(M0 + sr) * Kdim + sh;
  const __bf16* Wrow = W + (size_t)(N0 + sr) * Kdim + sh;

  v8f acc[8];
#pragma unroll
  for (int i = 0; i < 8; ++i) acc[i] = (v8f){0.f,0.f,0.f,0.f,0.f,0.f,0.f,0.f};

  auto stage = [&](int buf, int k0) {
#pragma unroll
    for (int c = 0; c < 4; ++c) {
      async_ld_b128((uint32_t)(size_t)&At[buf][sr][sh + c * 8],
                    (uint64_t)(size_t)(Arow + k0 + c * 8));
      async_ld_b128((uint32_t)(size_t)&Wt[buf][sr][sh + c * 8],
                    (uint64_t)(size_t)(Wrow + k0 + c * 8));
    }
  };

  stage(0, 0);

  int arow = wave * 16 + l16;
  int kbA  = hs * 16;
  int kbB  = hs * 16;

  for (int k0 = 0; k0 < Kdim; k0 += 64) {
    int buf = (k0 >> 6) & 1;
    wait_async0();        // our async stores into buf have landed in LDS
    __syncthreads();      // everyone's have
    if (k0 + 64 < Kdim) stage(buf ^ 1, k0 + 64);   // overlap next stage

#pragma unroll
    for (int kc = 0; kc < 2; ++kc) {
      int kb = kc * 32;
      v8bf a0 = *(const v8bf*)&At[buf][arow][kb + kbA];
      v8bf a1 = *(const v8bf*)&At[buf][arow][kb + kbA + 8];
      v16bf afrag = cat8(a0, a1);
      v16bf bfr[8];
#pragma unroll
      for (int nt = 0; nt < 8; ++nt)
        bfr[nt] = *(const v16bf*)&Wt[buf][nt * 16 + l16][kb + kbB];
#pragma unroll
      for (int nt = 0; nt < 8; ++nt)
        acc[nt] = wmma_bf16(afrag, bfr[nt], acc[nt]);
    }
  }

  // epilogue
#pragma unroll
  for (int nt = 0; nt < 8; ++nt) {
    int col = N0 + nt * 16 + l16;
    float bv = bias[col];
#pragma unroll
    for (int r = 0; r < 8; ++r) {
      int row = M0 + wave * 16 + r + hs * 8;
      float v = acc[nt][r] + bv;
      if (mode == 0) {
        int b = row / S_, s = row % S_;
        int h = col / D_, d = col % D_;
        outb[(((size_t)b * H_ + h) * S_ + s) * D_ + d] = (__bf16)v;
      } else {
        outf[(size_t)row * Ndim + col] = v;
      }
    }
  }
}

// ---------------------------------------------------------------- RoPE
__global__ void rope_kernel(__bf16* __restrict__ q, __bf16* __restrict__ k,
                            const float* __restrict__ cosT,
                            const float* __restrict__ sinT,
                            const int* __restrict__ ropeEx)
{
  size_t i = (size_t)blockIdx.x * blockDim.x + threadIdx.x;
  const size_t total = (size_t)B_ * H_ * S_ * (D_ / 2);
  if (i >= total) return;
  int d = (int)(i & 63);
  size_t r = i >> 6;                 // (b*H + h)*S + s
  int s = (int)(r % S_);
  size_t base = r * D_;
  if (s < ropeEx[0]) return;         // keep original below rope_exclude
  float c  = cosT[(size_t)s * D_ + d];
  float sn = sinT[(size_t)s * D_ + d];
  float q1 = (float)q[base + d], q2 = (float)q[base + d + 64];
  q[base + d]      = (__bf16)(q1 * c - q2 * sn);
  q[base + d + 64] = (__bf16)(q2 * c + q1 * sn);
  float k1 = (float)k[base + d], k2 = (float)k[base + d + 64];
  k[base + d]      = (__bf16)(k1 * c - k2 * sn);
  k[base + d + 64] = (__bf16)(k2 * c + k1 * sn);
}

// ---------------------------------------------------------------- flash attention
// grid: (S/128 query tiles, B*H). 8 waves; wave handles 16 query rows.
// K tile staged by the Tensor Data Mover; V transposed via registers.
__global__ __launch_bounds__(256) void flash_kernel(
    const __bf16* __restrict__ Qg, const __bf16* __restrict__ Kg,
    const __bf16* __restrict__ Vg, __bf16* __restrict__ ctx)
{
  __shared__ __align__(128) __bf16 Kt[128 * 128];   // [key][d]; reused as P buffer
  __shared__ __align__(128) __bf16 Vt[128 * 128];   // [d][key] (transposed)

  int t    = threadIdx.x;
  int lane = t & 31, wave = t >> 5;
  int l16  = lane & 15, hs = lane >> 4;
  int qt   = blockIdx.x;          // query tile
  int bh   = blockIdx.y;          // b*H + h
  const size_t headoff = (size_t)bh * S_ * D_;
  const __bf16* Qh = Qg + headoff;
  const __bf16* Kh = Kg + headoff;
  const __bf16* Vh = Vg + headoff;

  // Q fragments (A-layout) straight from global, kept in registers
  int qrow = qt * 128 + wave * 16 + l16;
  v16bf qf[4];
  {
    const __bf16* qp = Qh + (size_t)qrow * D_;
    int kb = hs * 16;
#pragma unroll
    for (int kc = 0; kc < 4; ++kc) {
      v8bf p0 = *(const v8bf*)(qp + kc * 32 + kb);
      v8bf p1 = *(const v8bf*)(qp + kc * 32 + kb + 8);
      qf[kc] = cat8(p0, p1);
    }
  }

  v8f O[8];
#pragma unroll
  for (int i = 0; i < 8; ++i) O[i] = (v8f){0.f,0.f,0.f,0.f,0.f,0.f,0.f,0.f};
  float mrun[8], lrun[8];
#pragma unroll
  for (int r = 0; r < 8; ++r) { mrun[r] = -3.0e38f; lrun[r] = 0.f; }

  __bf16* Pme = Kt + wave * 16 * 128;   // per-wave P slice (K tile dead by then)
  const int kbB = hs * 16;
  const uint32_t klds = (uint32_t)(size_t)&Kt[0];

  for (int kt = 0; kt <= qt; ++kt) {
    // ---- K tile via Tensor Data Mover (one wave issues; EXEC ignored by TDM)
    if (wave == 0) {
      uint64_t kga = (uint64_t)(size_t)(Kh + (size_t)(kt * 128) * D_);
      u32x4 g0 = { 1u,                                   // count=1 valid descriptor
                   klds,                                 // lds_addr
                   (uint32_t)kga,                        // global_addr[31:0]
                   (uint32_t)(((kga >> 32) & 0x01FFFFFFu) | 0x80000000u) }; // addr[56:32] | type=2
      u32x8 g1 = { 0x00010000u,     // workgroup_mask=0, data_size=2B
                   128u << 16,      // tensor_dim0[15:0]=128
                   128u << 16,      // tensor_dim0 hi=0, tensor_dim1[15:0]=128
                   128u << 16,      // tensor_dim1 hi=0, tile_dim0=128
                   128u,            // tile_dim1=128, tile_dim2=0
                   128u,            // tensor_dim0_stride=128
                   0u, 0u };        // stride hi / dim1_stride = 0
      asm volatile("tensor_load_to_lds %0, %1" :: "s"(g0), "s"(g1) : "memory");
      __builtin_amdgcn_s_wait_tensorcnt(0);
    }
    // ---- V tile transposed into LDS via registers (all threads)
#pragma unroll
    for (int it = 0; it < 4; ++it) {
      int chunk = it * 256 + t;                 // 0..1023
      int row = chunk >> 3;                     // key within tile
      int c16 = (chunk & 7) * 16;               // d offset
      v16bf vv = *(const v16bf*)&Vh[(size_t)(kt * 128 + row) * D_ + c16];
#pragma unroll
      for (int j = 0; j < 16; ++j) Vt[(c16 + j) * 128 + row] = vv[j];
    }
    __syncthreads();

    // ---- scores S = Q K^T (8 column tiles of 16 keys)
    v8f sc[8];
#pragma unroll
    for (int nt = 0; nt < 8; ++nt) {
      v8f a = (v8f){0.f,0.f,0.f,0.f,0.f,0.f,0.f,0.f};
#pragma unroll
      for (int kc = 0; kc < 4; ++kc) {
        v16bf bf = *(const v16bf*)&Kt[(nt * 16 + l16) * 128 + kc * 32 + kbB];
        a = wmma_bf16(qf[kc], bf, a);
      }
      sc[nt] = a;
    }
    __syncthreads();   // all waves done reading Kt before it is reused for P

    // ---- scale + causal mask
#pragma unroll
    for (int nt = 0; nt < 8; ++nt) {
      int kcol = kt * 128 + nt * 16 + l16;
#pragma unroll
      for (int r = 0; r < 8; ++r) {
        int row = qt * 128 + wave * 16 + r + hs * 8;
        float v = sc[nt][r] * SCALE_;
        sc[nt][r] = (kcol > row) ? -3.0e38f : v;
      }
    }

    // ---- online softmax: row max (across 8 tiles then across 16 lanes)
    float newm[8];
#pragma unroll
    for (int r = 0; r < 8; ++r) {
      float mx = sc[0][r];
#pragma unroll
      for (int nt = 1; nt < 8; ++nt) mx = fmaxf(mx, sc[nt][r]);
#pragma unroll
      for (int off = 1; off < 16; off <<= 1)
        mx = fmaxf(mx, __shfl_xor(mx, off, 32));
      newm[r] = fmaxf(mrun[r], mx);
    }

    // ---- P = exp(S - newm): row sums, spill P (bf16) to per-wave LDS slice
    float rsum[8];
#pragma unroll
    for (int r = 0; r < 8; ++r) rsum[r] = 0.f;
#pragma unroll
    for (int nt = 0; nt < 8; ++nt) {
#pragma unroll
      for (int r = 0; r < 8; ++r) {
        float p = __expf(sc[nt][r] - newm[r]);
        rsum[r] += p;
        Pme[(r + hs * 8) * 128 + nt * 16 + l16] = (__bf16)p;
      }
    }
#pragma unroll
    for (int r = 0; r < 8; ++r) {
      float s8 = rsum[r];
#pragma unroll
      for (int off = 1; off < 16; off <<= 1) s8 += __shfl_xor(s8, off, 32);
      float corr = __expf(mrun[r] - newm[r]);
      lrun[r] = lrun[r] * corr + s8;
      mrun[r] = newm[r];
      rsum[r] = corr;                 // reuse as O correction factor
    }
#pragma unroll
    for (int dt = 0; dt < 8; ++dt)
#pragma unroll
      for (int r = 0; r < 8; ++r) O[dt][r] = O[dt][r] * rsum[r];

    // ---- our LDS P stores must land before A-layout reads (same wave only)
    asm volatile("s_wait_dscnt 0x0" ::: "memory");
    v16bf pf[4];
    {
      const __bf16* pp = Pme + l16 * 128;
      int kb = hs * 16;
#pragma unroll
      for (int kc = 0; kc < 4; ++kc) {
        v8bf p0 = *(const v8bf*)(pp + kc * 32 + kb);
        v8bf p1 = *(const v8bf*)(pp + kc * 32 + kb + 8);
        pf[kc] = cat8(p0, p1);
      }
    }

    // ---- O += P @ V
#pragma unroll
    for (int dt = 0; dt < 8; ++dt) {
      v8f a = O[dt];
#pragma unroll
      for (int kc = 0; kc < 4; ++kc) {
        v16bf bf = *(const v16bf*)&Vt[(dt * 16 + l16) * 128 + kc * 32 + kbB];
        a = wmma_bf16(pf[kc], bf, a);
      }
      O[dt] = a;
    }
    __syncthreads();   // P/Vt reads done before next staging overwrites
  }

  // ---- normalize and store ctx as [B, S, H*D] (ready for O-projection)
  int b = bh >> 4, h = bh & 15;
#pragma unroll
  for (int dt = 0; dt < 8; ++dt) {
    int col = dt * 16 + l16;
#pragma unroll
    for (int r = 0; r < 8; ++r) {
      int row = qt * 128 + wave * 16 + r + hs * 8;   // s
      float o = O[dt][r] / lrun[r];
      ctx[((size_t)b * S_ + row) * E_ + h * D_ + col] = (__bf16)o;
    }
  }
}

// ---------------------------------------------------------------- launcher
extern "C" void kernel_launch(void* const* d_in, const int* in_sizes, int n_in,
                              void* d_out, int out_size, void* d_ws, size_t ws_size,
                              hipStream_t stream)
{
  (void)in_sizes; (void)n_in; (void)out_size; (void)ws_size;
  const float* x    = (const float*)d_in[0];
  /* d_in[1] = mask: causal, applied analytically */
  const float* cosT = (const float*)d_in[2];
  const float* sinT = (const float*)d_in[3];
  const float* Wq   = (const float*)d_in[4];
  const float* bq   = (const float*)d_in[5];
  const float* Wk   = (const float*)d_in[6];
  const float* bk   = (const float*)d_in[7];
  const float* Wv   = (const float*)d_in[8];
  const float* bv   = (const float*)d_in[9];
  const float* Wo   = (const float*)d_in[10];
  const float* bo   = (const float*)d_in[11];
  const int* ropeEx = (const int*)d_in[12];

  const size_t nX = (size_t)B_ * S_ * E_;   // 8,388,608
  const size_t nW = (size_t)E_ * E_;        // 4,194,304

  __bf16* p   = (__bf16*)d_ws;
  __bf16* Xb  = p; p += nX;
  __bf16* Wqb = p; p += nW;
  __bf16* Wkb = p; p += nW;
  __bf16* Wvb = p; p += nW;
  __bf16* Wob = p; p += nW;
  __bf16* Qb  = p; p += nX;
  __bf16* Kb  = p; p += nX;
  __bf16* Vb  = p; p += nX;
  __bf16* Cb  = p; p += nX;

  // 1) convert inputs to bf16 (vectorized x4)
  cvt_f32_bf16x4<<<(int)((nX / 4 + 255) / 256), 256, 0, stream>>>(
      (const float4*)x, (uint2*)Xb, (int)(nX / 4));
  cvt_f32_bf16x4<<<(int)((nW / 4 + 255) / 256), 256, 0, stream>>>(
      (const float4*)Wq, (uint2*)Wqb, (int)(nW / 4));
  cvt_f32_bf16x4<<<(int)((nW / 4 + 255) / 256), 256, 0, stream>>>(
      (const float4*)Wk, (uint2*)Wkb, (int)(nW / 4));
  cvt_f32_bf16x4<<<(int)((nW / 4 + 255) / 256), 256, 0, stream>>>(
      (const float4*)Wv, (uint2*)Wvb, (int)(nW / 4));
  cvt_f32_bf16x4<<<(int)((nW / 4 + 255) / 256), 256, 0, stream>>>(
      (const float4*)Wo, (uint2*)Wob, (int)(nW / 4));

  // 2) QKV projections (WMMA GEMM), scatter to [B,H,S,D]
  dim3 ggrid(E_ / 128, M_ / 128);   // (16, 32)
  gemm_bf16<<<ggrid, 256, 0, stream>>>(Xb, Wqb, bq, Qb, nullptr, 0);
  gemm_bf16<<<ggrid, 256, 0, stream>>>(Xb, Wkb, bk, Kb, nullptr, 0);
  gemm_bf16<<<ggrid, 256, 0, stream>>>(Xb, Wvb, bv, Vb, nullptr, 0);

  // 3) RoPE on Q and K
  {
    size_t total = (size_t)B_ * H_ * S_ * (D_ / 2);
    rope_kernel<<<(int)((total + 255) / 256), 256, 0, stream>>>(Qb, Kb, cosT, sinT, ropeEx);
  }

  // 4) causal flash attention -> ctx [B,S,E] bf16
  flash_kernel<<<dim3(S_ / 128, B_ * H_), 256, 0, stream>>>(Qb, Kb, Vb, Cb);

  // 5) output projection -> f32 d_out
  gemm_bf16<<<ggrid, 256, 0, stream>>>(Cb, Wob, bo, nullptr, (float*)d_out, 1);
}
